// ScaledQuantizedLinearINT4_300647711290
// MI455X (gfx1250) — 8-bit matrix operand order repaired and checked by reading, compiled, not run on hardware
//
#include <hip/hip_runtime.h>
#include <math.h>

typedef __attribute__((ext_vector_type(8))) int v8i;

__device__ __forceinline__ float powlaw(float t, float a) {
  return copysignf(powf(fabsf(t), a), t);
}

__global__ void wminmax_kernel(const float* __restrict__ w, int n4,
                               float* __restrict__ partial) {
  __shared__ float smn[256];
  __shared__ float smx[256];
  float mn = 3.4e38f, mx = -3.4e38f;
  const float4* w4 = (const float4*)w;
  for (int i = blockIdx.x * blockDim.x + threadIdx.x; i < n4;
       i += gridDim.x * blockDim.x) {
    float4 v = w4[i];
    mn = fminf(mn, fminf(fminf(v.x, v.y), fminf(v.z, v.w)));
    mx = fmaxf(mx, fmaxf(fmaxf(v.x, v.y), fmaxf(v.z, v.w)));
  }
  smn[threadIdx.x] = mn;
  smx[threadIdx.x] = mx;
  __syncthreads();
  for (int s = 128; s > 0; s >>= 1) {
    if ((int)threadIdx.x < s) {
      smn[threadIdx.x] = fminf(smn[threadIdx.x], smn[threadIdx.x + s]);
      smx[threadIdx.x] = fmaxf(smx[threadIdx.x], smx[threadIdx.x + s]);
    }
    __syncthreads();
  }
  if (threadIdx.x == 0) {
    partial[2 * blockIdx.x + 0] = smn[0];
    partial[2 * blockIdx.x + 1] = smx[0];
  }
}

__global__ void finalize_kernel(const float* __restrict__ partial, int nparts,
                                const float* __restrict__ alpha_p,
                                float* __restrict__ fin) {
  __shared__ float smn[256];
  __shared__ float smx[256];
  float mn = 3.4e38f, mx = -3.4e38f;
  for (int i = threadIdx.x; i < nparts; i += blockDim.x) {
    mn = fminf(mn, partial[2 * i + 0]);
    mx = fmaxf(mx, partial[2 * i + 1]);
  }
  smn[threadIdx.x] = mn;
  smx[threadIdx.x] = mx;
  __syncthreads();
  for (int s = 128; s > 0; s >>= 1) {
    if ((int)threadIdx.x < s) {
      smn[threadIdx.x] = fminf(smn[threadIdx.x], smn[threadIdx.x + s]);
      smx[threadIdx.x] = fmaxf(smx[threadIdx.x], smx[threadIdx.x + s]);
    }
    __syncthreads();
  }
  if (threadIdx.x == 0) {
    float a = alpha_p[0];
    float mns = powlaw(smn[0], a);
    float mxs = powlaw(smx[0], a);
    fin[0] = mns;
    fin[1] = (mxs - mns) / 15.0f;
  }
}

__global__ void quant_w_kernel(const float* __restrict__ w,
                               unsigned char* __restrict__ wq,
                               const float* __restrict__ fin,
                               const float* __restrict__ alpha_p, int n4) {
  int i = blockIdx.x * blockDim.x + threadIdx.x;
  if (i >= n4) return;
  float a = alpha_p[0];
  float mn = fin[0];
  float inv_sc = 1.0f / fin[1];
  float4 v = ((const float4*)w)[i];
  uchar4 q;
  q.x = (unsigned char)(int)fminf(fmaxf(rintf((powlaw(v.x, a) - mn) * inv_sc), 0.0f), 15.0f);
  q.y = (unsigned char)(int)fminf(fmaxf(rintf((powlaw(v.y, a) - mn) * inv_sc), 0.0f), 15.0f);
  q.z = (unsigned char)(int)fminf(fmaxf(rintf((powlaw(v.z, a) - mn) * inv_sc), 0.0f), 15.0f);
  q.w = (unsigned char)(int)fminf(fmaxf(rintf((powlaw(v.w, a) - mn) * inv_sc), 0.0f), 15.0f);
  ((uchar4*)wq)[i] = q;
}

__global__ void quant_x_kernel(const float* __restrict__ x,
                               signed char* __restrict__ xq,
                               const float* __restrict__ alpha_p, int n4) {
  int i = blockIdx.x * blockDim.x + threadIdx.x;
  if (i >= n4) return;
  float a = alpha_p[0];
  float4 v = ((const float4*)x)[i];
  char4 q;
  if (a == 0.5f) {
    q.x = (signed char)(int)copysignf(sqrtf(fabsf(v.x)), v.x);
    q.y = (signed char)(int)copysignf(sqrtf(fabsf(v.y)), v.y);
    q.z = (signed char)(int)copysignf(sqrtf(fabsf(v.z)), v.z);
    q.w = (signed char)(int)copysignf(sqrtf(fabsf(v.w)), v.w);
  } else {
    q.x = (signed char)(int)powlaw(v.x, a);
    q.y = (signed char)(int)powlaw(v.y, a);
    q.z = (signed char)(int)powlaw(v.z, a);
    q.w = (signed char)(int)powlaw(v.w, a);
  }
  ((char4*)xq)[i] = q;
}

__device__ __forceinline__ void compute_ktile(const char* __restrict__ As,
                                              const char* __restrict__ Bs,
                                              int wm, int wn, int half,
                                              int l16, v8i acc[4][2]) {
  v8i bfr[2];
#pragma unroll
  for (int bt = 0; bt < 2; bt++) {
    const char* bb = Bs + (wn * 32 + bt * 16 + l16) * 64 + half * 16;
    int4 e0 = *(const int4*)(bb + 0);
    int4 e1 = *(const int4*)(bb + 32);
    bfr[bt] = (v8i){e0.x, e0.y, e0.z, e0.w, e1.x, e1.y, e1.z, e1.w};
  }
  v8i afr[4];
#pragma unroll
  for (int mt = 0; mt < 4; mt++) {
    const char* ab = As + (wm * 64 + mt * 16 + l16) * 64 + half * 16;
    int2 d0 = *(const int2*)(ab + 0);
    int2 d1 = *(const int2*)(ab + 8);
    int2 d2 = *(const int2*)(ab + 32);
    int2 d3 = *(const int2*)(ab + 40);
    afr[mt] = (v8i){d0.x, d0.y, d1.x, d1.y, d2.x, d2.y, d3.x, d3.y};
  }
#pragma unroll
  for (int mt = 0; mt < 4; mt++) {
#pragma unroll
    for (int bt = 0; bt < 2; bt++) {
      acc[mt][bt] = __builtin_amdgcn_wmma_i32_16x16x64_iu8(
          true, afr[mt], false, bfr[bt], acc[mt][bt], false, false);
    }
  }
}

__global__ __launch_bounds__(256) void gemm_iu8_kernel(
    const signed char* __restrict__ Aq, const unsigned char* __restrict__ Bq,
    float* __restrict__ out, const float* __restrict__ alpha_p, int M, int N,
    int K) {
  __shared__ __align__(16) char As[128 * 64];
  __shared__ __align__(16) char Bs[128 * 64];

  const int tid = threadIdx.x;
  const int lane = tid & 31;
  const int wave = tid >> 5;
  const int wm = wave & 1;
  const int wn = wave >> 1;
  const int half = lane >> 4;
  const int l16 = lane & 15;
  const int m0 = blockIdx.y * 128;
  const int n0 = blockIdx.x * 128;

  const int crow = tid >> 1;
  const int coff = (tid & 1) * 32;
  const char* gA = (const char*)Aq + (size_t)(m0 + crow) * K + coff;
  const char* gB = (const char*)Bq + (size_t)(n0 + crow) * K + coff;
  char* sA = As + crow * 64 + coff;
  char* sB = Bs + crow * 64 + coff;

  *(int4*)(sA + 0)  = *(const int4*)(gA + 0);
  *(int4*)(sA + 16) = *(const int4*)(gA + 16);
  *(int4*)(sB + 0)  = *(const int4*)(gB + 0);
  *(int4*)(sB + 16) = *(const int4*)(gB + 16);
  __syncthreads();

  const v8i vzero = {0, 0, 0, 0, 0, 0, 0, 0};
  v8i acc[4][2];
#pragma unroll
  for (int i = 0; i < 4; i++)
#pragma unroll
    for (int j = 0; j < 2; j++) acc[i][j] = vzero;

  const int KT = K >> 6;
  for (int kt = 0; kt < KT - 1; ++kt) {
    const int knext = (kt + 1) << 6;
    int4 pa0 = *(const int4*)(gA + knext + 0);
    int4 pa1 = *(const int4*)(gA + knext + 16);
    int4 pb0 = *(const int4*)(gB + knext + 0);
    int4 pb1 = *(const int4*)(gB + knext + 16);

    compute_ktile(As, Bs, wm, wn, half, l16, acc);

    __syncthreads();
    *(int4*)(sA + 0)  = pa0;
    *(int4*)(sA + 16) = pa1;
    *(int4*)(sB + 0)  = pb0;
    *(int4*)(sB + 16) = pb1;
    __syncthreads();
  }
  compute_ktile(As, Bs, wm, wn, half, l16, acc);

  const float a = alpha_p[0];
  const int gcol_base = n0 + wn * 32 + l16;
  if (a == 0.5f) {
#pragma unroll
    for (int mt = 0; mt < 4; mt++) {
#pragma unroll
      for (int bt = 0; bt < 2; bt++) {
#pragma unroll
        for (int r = 0; r < 8; r++) {
          const int grow = m0 + wm * 64 + mt * 16 + half * 8 + r;
          const float v = (float)acc[mt][bt][r];
          out[(size_t)grow * N + gcol_base + bt * 16] = v * fabsf(v);
        }
      }
    }
  } else {
    const float inv_a = 1.0f / a;
#pragma unroll
    for (int mt = 0; mt < 4; mt++) {
#pragma unroll
      for (int bt = 0; bt < 2; bt++) {
#pragma unroll
        for (int r = 0; r < 8; r++) {
          const int grow = m0 + wm * 64 + mt * 16 + half * 8 + r;
          const float v = (float)acc[mt][bt][r];
          out[(size_t)grow * N + gcol_base + bt * 16] =
              copysignf(powf(fabsf(v), inv_a), v);
        }
      }
    }
  }
}

extern "C" void kernel_launch(void* const* d_in, const int* in_sizes, int n_in,
                              void* d_out, int out_size, void* d_ws,
                              size_t ws_size, hipStream_t stream) {
  const float* x = (const float*)d_in[0];
  const float* w = (const float*)d_in[1];
  const float* alpha = (const float*)d_in[2];

  const int K = 4096;
  const int N = 4096;
  const int M = in_sizes[0] / K;

  char* ws = (char*)d_ws;
  signed char* xq = (signed char*)ws;
  unsigned char* wq = (unsigned char*)(ws + (size_t)M * K);
  float* partial = (float*)(ws + (size_t)M * K + (size_t)N * K);
  float* fin = partial + 512;

  const int wn4 = (N * K) / 4;
  const int xn4 = (M * K) / 4;

  wminmax_kernel<<<256, 256, 0, stream>>>(w, wn4, partial);
  finalize_kernel<<<1, 256, 0, stream>>>(partial, 256, alpha, fin);
  quant_w_kernel<<<(wn4 + 255) / 256, 256, 0, stream>>>(w, wq, fin, alpha, wn4);
  quant_x_kernel<<<(xn4 + 255) / 256, 256, 0, stream>>>(x, xq, alpha, xn4);

  dim3 grid(N / 128, M / 128);
  gemm_iu8_kernel<<<grid, 256, 0, stream>>>(xq, wq, (float*)d_out, alpha, M, N,
                                            K);
}
